// ConvLSTM_16286515986723
// MI455X (gfx1250) — matrix operand order repaired and checked by reading, not compiled, not run on hardware
//
#include <hip/hip_runtime.h>
#include <hip/hip_bf16.h>
#include <math.h>

// ---------------------------------------------------------------------------
// ConvLSTM (3 layers) on gfx1250 via implicit-GEMM WMMA bf16 convolutions.
//   B=8, T=16, H=W=28, CIN=192, HCS={64,32,64}, K=3 SAME.
// GEMM K-order: k = r*CI + ci (r = ky*3+kx); CI multiple of 32 so each 32-wide
// K chunk has a fixed conv tap + 32 consecutive channels. Activations live in
// bf16 zero-padded pixel-major [b][30][30][CI] buffers, so A-fragment loads
// need no bounds checks and all tap offsets are instruction immediates.
// Each wave computes a 32M x 64N tile; per chunk: 4 A loads + 8 B loads feed
// 8 v_wmma ops, with all 4 B tiles live simultaneously for deep pipelining.
// ---------------------------------------------------------------------------

typedef __attribute__((ext_vector_type(16))) __bf16 v16bf;
typedef __attribute__((ext_vector_type(8)))  __bf16 v8bf;
typedef __attribute__((ext_vector_type(8)))  float  v8f;

#define HH 28
#define WW 28
#define PIX 784              // 28*28 interior pixels
#define PW  30               // padded width/height
#define PPIX 900             // 30*30 padded pixels
#define NB 8                 // batch
#define NT_STEPS 16          // timesteps
#define MROWS (NB * PIX)     // 6272 GEMM rows (divisible by 64 and 128)

// ---------------- weight packing -------------------------------------------
// W[(o, ci, ky, kx)] f32 -> bf16 B fragments, K order k = r*I + ci.
// B operand of V_WMMA_F32_16X16X32_BF16 (32x16 tile):
//   lanes 0-15  : K = kc*32 + {0..7, 16..23},  col n = lane
//   lanes 16-31 : K = kc*32 + {8..15, 24..31}, col n = lane-16
// packed[((kc*NT + ntile)*32 + lane)*16 + j], 32B contiguous per lane.
__global__ void pack_weights(const float* __restrict__ W, __bf16* __restrict__ dst,
                             int I, int O, int n) {
    int idx = blockIdx.x * blockDim.x + threadIdx.x;
    if (idx >= n) return;
    int o  = idx / (I * 9);
    int rm = idx - o * (I * 9);
    int ci = rm / 9;
    int r  = rm - ci * 9;
    int k  = r * I + ci;
    int kc = k >> 5;
    int kl = k & 31;
    int ntile = o >> 4;
    int nl    = o & 15;
    int lane  = (((kl >> 3) & 1) << 4) + nl;
    int j     = (kl & 7) + ((kl >> 4) << 3);
    int NT    = O >> 4;
    dst[((size_t)(kc * NT + ntile) * 32 + lane) * 16 + j] = (__bf16)W[idx];
}

__global__ void zero_u32(unsigned int* __restrict__ p, int n) {
    int i = blockIdx.x * blockDim.x + threadIdx.x;
    if (i < n) p[i] = 0u;
}

// x slice (B,CIN,28,28) f32 NCHW -> bf16 padded pixel-major [b][30][30][CI]
__global__ void xform_x(const float* __restrict__ xt, size_t bstride,
                        __bf16* __restrict__ dst, int CI, int n) {
    int idx = blockIdx.x * blockDim.x + threadIdx.x;
    if (idx >= n) return;
    int ci = idx % CI;
    int rp = idx / CI;
    int p  = rp % PIX;
    int b  = rp / PIX;
    int y  = p / WW;
    int x  = p - y * WW;
    int pp = (y + 1) * PW + (x + 1);          // padded pixel index
    dst[((size_t)b * PPIX + pp) * CI + ci] = (__bf16)xt[(size_t)b * bstride + (size_t)ci * PIX + p];
}

// ---------------- dual-source implicit GEMM conv ---------------------------
// ap0/ap1: padded center pixels of the two 16-row A subtiles (+laneHi*8).
template<int CI, int CO>
__device__ __forceinline__ void k_loop2(const __bf16* __restrict__ ap0,
                                        const __bf16* __restrict__ ap1,
                                        const __bf16* __restrict__ bp_lane,
                                        v8f acc0[4], v8f acc1[4]) {
    const int NT = CO >> 4;
    const int CB = CI >> 5;                   // chunks per tap
#pragma unroll
    for (int r = 0; r < 9; ++r) {
        const int dy  = r / 3 - 1;
        const int dx  = r % 3 - 1;
        const int off = (dy * PW + dx) * CI;  // compile-time tap offset
#pragma unroll
        for (int cb = 0; cb < CB; ++cb) {
            // A fragments (two 16B loads each, contiguous channels)
            v8bf lo0 = *(const v8bf*)(ap0 + off + cb * 32);
            v8bf hi0 = *(const v8bf*)(ap0 + off + cb * 32 + 16);
            v8bf lo1 = *(const v8bf*)(ap1 + off + cb * 32);
            v8bf hi1 = *(const v8bf*)(ap1 + off + cb * 32 + 16);
            v16bf a0 = __builtin_shufflevector(lo0, hi0,
                    0, 1, 2, 3, 4, 5, 6, 7, 8, 9, 10, 11, 12, 13, 14, 15);
            v16bf a1 = __builtin_shufflevector(lo1, hi1,
                    0, 1, 2, 3, 4, 5, 6, 7, 8, 9, 10, 11, 12, 13, 14, 15);
            // All four B tiles live at once -> one load clause, staggered waits
            const __bf16* bp = bp_lane + (size_t)((r * CB + cb) * NT) * 512;
            v16bf b0 = *(const v16bf*)(bp);
            v16bf b1 = *(const v16bf*)(bp + 512);
            v16bf b2 = *(const v16bf*)(bp + 1024);
            v16bf b3 = *(const v16bf*)(bp + 1536);
            acc0[0] = __builtin_amdgcn_wmma_f32_16x16x32_bf16(
                false, a0, false, b0, (short)0, acc0[0], false, false);
            acc1[0] = __builtin_amdgcn_wmma_f32_16x16x32_bf16(
                false, a1, false, b0, (short)0, acc1[0], false, false);
            acc0[1] = __builtin_amdgcn_wmma_f32_16x16x32_bf16(
                false, a0, false, b1, (short)0, acc0[1], false, false);
            acc1[1] = __builtin_amdgcn_wmma_f32_16x16x32_bf16(
                false, a1, false, b1, (short)0, acc1[1], false, false);
            acc0[2] = __builtin_amdgcn_wmma_f32_16x16x32_bf16(
                false, a0, false, b2, (short)0, acc0[2], false, false);
            acc1[2] = __builtin_amdgcn_wmma_f32_16x16x32_bf16(
                false, a1, false, b2, (short)0, acc1[2], false, false);
            acc0[3] = __builtin_amdgcn_wmma_f32_16x16x32_bf16(
                false, a0, false, b3, (short)0, acc0[3], false, false);
            acc1[3] = __builtin_amdgcn_wmma_f32_16x16x32_bf16(
                false, a1, false, b3, (short)0, acc1[3], false, false);
        }
    }
}

template<int CI0, int CI1, int CO, int WM, int WN>
__global__ __launch_bounds__(WM * WN * 32)
void conv_dual(const __bf16* __restrict__ A0,   // [b][30][30][CI0] bf16 padded
               const __bf16* __restrict__ A1,   // [b][30][30][CI1] bf16 padded
               const __bf16* __restrict__ Bp0, const __bf16* __restrict__ Bp1,
               float* __restrict__ G) {
    const int lane   = threadIdx.x & 31;
    const int wave   = threadIdx.x >> 5;
    const int wm     = wave / WN;
    const int wn     = wave - wm * WN;
    const int m_base = blockIdx.x * (WM * 32) + wm * 32;   // 32 rows per wave
    const int n_base = wn * 64;
    const int laneHi = lane >> 4;

    // Two A subtile rows for this lane (may straddle a batch image boundary:
    // 784 % 32 != 0, so compute (b, pixel) independently per subtile).
    const int mrow0 = m_base + (lane & 15);
    const int mrow1 = mrow0 + 16;
    const int b0 = mrow0 / PIX;
    const int p0 = mrow0 - b0 * PIX;
    const int b1 = mrow1 / PIX;
    const int p1 = mrow1 - b1 * PIX;
    const int y0 = p0 / WW, x0 = p0 - y0 * WW;
    const int y1 = p1 / WW, x1 = p1 - y1 * WW;
    const int pp0 = (y0 + 1) * PW + (x0 + 1);
    const int pp1 = (y1 + 1) * PW + (x1 + 1);

    v8f acc0[4], acc1[4];
#pragma unroll
    for (int t = 0; t < 4; ++t) { acc0[t] = {}; acc1[t] = {}; }

    {
        const __bf16* a0 = A0 + ((size_t)b0 * PPIX + pp0) * CI0 + laneHi * 8;
        const __bf16* a1 = A0 + ((size_t)b1 * PPIX + pp1) * CI0 + laneHi * 8;
        const __bf16* bl = Bp0 + (size_t)(n_base >> 4) * 512 + (size_t)lane * 16;
        k_loop2<CI0, CO>(a0, a1, bl, acc0, acc1);
    }
    {
        const __bf16* a0 = A1 + ((size_t)b0 * PPIX + pp0) * CI1 + laneHi * 8;
        const __bf16* a1 = A1 + ((size_t)b1 * PPIX + pp1) * CI1 + laneHi * 8;
        const __bf16* bl = Bp1 + (size_t)(n_base >> 4) * 512 + (size_t)lane * 16;
        k_loop2<CI1, CO>(a0, a1, bl, acc0, acc1);
    }

    // C/D layout: VGPR r -> row (r + laneHi*8), col = n_base + t*16 + (lane&15)
#pragma unroll
    for (int t = 0; t < 4; ++t) {
        int co = n_base + t * 16 + (lane & 15);
#pragma unroll
        for (int r = 0; r < 8; ++r) {
            int m0 = m_base + laneHi * 8 + r;
            G[(size_t)m0 * CO + co] = acc0[t][r];
            int m1 = m0 + 16;
            G[(size_t)m1 * CO + co] = acc1[t][r];
        }
    }
}

// ---------------- LSTM gate update -----------------------------------------
// Channel-minor indexing: idx = (b*PIX + p)*Hc + hc (coalesced G reads).
// Writes next conv input directly as bf16 padded pixel-major; c stays f32.
__global__ void gate_kernel(const float* __restrict__ G,
                            const float* __restrict__ bx,
                            const float* __restrict__ Wp,
                            __bf16* __restrict__ Hbf, float* __restrict__ C,
                            float* __restrict__ Out,
                            int Hc, int t, int n) {
    int idx = blockIdx.x * blockDim.x + threadIdx.x;
    if (idx >= n) return;
    int hc = idx % Hc;
    int rp = idx / Hc;
    int p  = rp % PIX;
    int b  = rp / PIX;
    int y  = p / WW;
    int x  = p - y * WW;
    int pp = (y + 1) * PW + (x + 1);
    int CO = 4 * Hc;
    const float* g = G + (size_t)(b * PIX + p) * CO;
    float gi = g[hc]          + bx[hc];
    float gf = g[Hc + hc]     + bx[Hc + hc];
    float gc = g[2 * Hc + hc] + bx[2 * Hc + hc];
    float go = g[3 * Hc + hc] + bx[3 * Hc + hc];
    float pi = Wp[((size_t)0 * Hc + hc) * PIX + p];
    float pf = Wp[((size_t)1 * Hc + hc) * PIX + p];
    float po = Wp[((size_t)2 * Hc + hc) * PIX + p];
    float c  = C[idx];
    float i  = 1.0f / (1.0f + __expf(-(gi + c * pi)));
    float f  = 1.0f / (1.0f + __expf(-(gf + c * pf)));
    float cc = f * c + i * tanhf(gc);
    float o  = 1.0f / (1.0f + __expf(-(go + cc * po)));
    float h  = o * tanhf(cc);
    C[idx] = cc;
    Hbf[((size_t)b * PPIX + pp) * Hc + hc] = (__bf16)h;
    if (Out)
        Out[(((size_t)b * NT_STEPS + t) * 64 + hc) * PIX + p] = h;
}

// ---------------- workspace layout (bytes, 256-aligned) --------------------
#define OFF_PWX0  0u            // 1728*256*2 = 884736
#define OFF_PWH0  884736u       //  576*256*2 = 294912
#define OFF_PWX1  1179648u      //  576*128*2 = 147456
#define OFF_PWH1  1327104u      //  288*128*2 =  73728
#define OFF_PWX2  1400832u      //  288*256*2 = 147456
#define OFF_PWH2  1548288u      //  576*256*2 = 294912
#define OFF_XBF   1843200u      // 8*900*192*2 = 2764800 (padded)
#define OFF_HBF0  4608000u      // 8*900*64*2  =  921600
#define OFF_HBF1  5529600u      // 8*900*32*2  =  460800
#define OFF_HBF2  5990400u      // 8*900*64*2  =  921600
#define OFF_C0    6912000u      // 8*784*64*4  = 1605632
#define OFF_C1    8517632u      // 8*784*32*4  =  802816
#define OFF_C2    9320448u      // 8*784*64*4  = 1605632
#define OFF_G     10926080u     // 6272*256*4  = 6422528
// zero region: XBF..C2 = [1843200, 10926080) = 9082880 B = 2270720 u32

extern "C" void kernel_launch(void* const* d_in, const int* in_sizes, int n_in,
                              void* d_out, int out_size, void* d_ws, size_t ws_size,
                              hipStream_t stream) {
    const float* x   = (const float*)d_in[0];
    const float* Wx0 = (const float*)d_in[1];
    const float* bx0 = (const float*)d_in[2];
    const float* Wh0 = (const float*)d_in[3];
    const float* Wp0 = (const float*)d_in[4];
    const float* Wx1 = (const float*)d_in[5];
    const float* bx1 = (const float*)d_in[6];
    const float* Wh1 = (const float*)d_in[7];
    const float* Wp1 = (const float*)d_in[8];
    const float* Wx2 = (const float*)d_in[9];
    const float* bx2 = (const float*)d_in[10];
    const float* Wh2 = (const float*)d_in[11];
    const float* Wp2 = (const float*)d_in[12];
    float* out = (float*)d_out;

    char* ws = (char*)d_ws;
    __bf16* pWx0 = (__bf16*)(ws + OFF_PWX0);
    __bf16* pWh0 = (__bf16*)(ws + OFF_PWH0);
    __bf16* pWx1 = (__bf16*)(ws + OFF_PWX1);
    __bf16* pWh1 = (__bf16*)(ws + OFF_PWH1);
    __bf16* pWx2 = (__bf16*)(ws + OFF_PWX2);
    __bf16* pWh2 = (__bf16*)(ws + OFF_PWH2);
    __bf16* Xbf  = (__bf16*)(ws + OFF_XBF);
    __bf16* Hbf0 = (__bf16*)(ws + OFF_HBF0);
    __bf16* Hbf1 = (__bf16*)(ws + OFF_HBF1);
    __bf16* Hbf2 = (__bf16*)(ws + OFF_HBF2);
    float*  C0   = (float*)(ws + OFF_C0);
    float*  C1   = (float*)(ws + OFF_C1);
    float*  C2   = (float*)(ws + OFF_C2);
    float*  G    = (float*)(ws + OFF_G);

    // --- pack weights to bf16 fragments ---
    auto packN = [&](const float* W, __bf16* dst, int I, int O) {
        int n = O * I * 9;
        pack_weights<<<(n + 255) / 256, 256, 0, stream>>>(W, dst, I, O, n);
    };
    packN(Wx0, pWx0, 192, 256);
    packN(Wh0, pWh0,  64, 256);
    packN(Wx1, pWx1,  64, 128);
    packN(Wh1, pWh1,  32, 128);
    packN(Wx2, pWx2,  32, 256);
    packN(Wh2, pWh2,  64, 256);

    // --- zero padded activation buffers (halo stays zero forever) + c ---
    {
        int nz = 2270720;
        zero_u32<<<(nz + 255) / 256, 256, 0, stream>>>((unsigned int*)Xbf, nz);
    }

    const size_t xTStride = (size_t)192 * PIX;            // per-timestep
    const size_t xBStride = (size_t)NT_STEPS * 192 * PIX; // per-batch

    for (int t = 0; t < NT_STEPS; ++t) {
        // x_t -> bf16 padded pixel-major
        {
            int n = NB * PIX * 192;
            xform_x<<<(n + 255) / 256, 256, 0, stream>>>(
                x + (size_t)t * xTStride, xBStride, Xbf, 192, n);
        }

        // Layer 0: gates = conv(x_t, Wx0) + conv(h0, Wh0)
        conv_dual<192, 64, 256, 2, 4><<<MROWS / 64, 256, 0, stream>>>(
            Xbf, Hbf0, pWx0, pWh0, G);
        gate_kernel<<<(NB * 64 * PIX + 255) / 256, 256, 0, stream>>>(
            G, bx0, Wp0, Hbf0, C0, nullptr, 64, t, NB * 64 * PIX);

        // Layer 1: gates = conv(h0, Wx1) + conv(h1, Wh1)
        conv_dual<64, 32, 128, 4, 2><<<MROWS / 128, 256, 0, stream>>>(
            Hbf0, Hbf1, pWx1, pWh1, G);
        gate_kernel<<<(NB * 32 * PIX + 255) / 256, 256, 0, stream>>>(
            G, bx1, Wp1, Hbf1, C1, nullptr, 32, t, NB * 32 * PIX);

        // Layer 2: gates = conv(h1, Wx2) + conv(h2, Wh2); writes output slice
        conv_dual<32, 64, 256, 2, 4><<<MROWS / 64, 256, 0, stream>>>(
            Hbf1, Hbf2, pWx2, pWh2, G);
        gate_kernel<<<(NB * 64 * PIX + 255) / 256, 256, 0, stream>>>(
            G, bx2, Wp2, Hbf2, C2, out, 64, t, NB * 64 * PIX);
    }
}
